// GlobalAttentionBlock_38268158607446
// MI455X (gfx1250) — matrix operand order repaired and checked by reading, not compiled, not run on hardware
//
#include <hip/hip_runtime.h>
#include <hip/hip_bf16.h>

// ---------------------------------------------------------------------------
// CDNA5 (gfx1250) implementation of a SAM-style global-attention ViT block.
// All matmuls run through v_wmma_f32_16x16x32_f16 (wave32 WMMA) with
// LDS-staged, double-buffered operand tiles moved by the gfx1250 async
// global->LDS path (global_load_async_to_lds_b128 + s_wait_asynccnt).
// ---------------------------------------------------------------------------

typedef __attribute__((ext_vector_type(16))) _Float16 v16h;
typedef __attribute__((ext_vector_type(8)))  _Float16 v8h;
typedef __attribute__((ext_vector_type(8)))  float    v8f;
typedef __attribute__((ext_vector_type(4)))  int      v4i;

#define TOKENS 2304     // 48*48
#define CDIM   768
#define HEADS  12
#define HD     64
#define GRID_HW 48
#define MLPH   3072

#define LDS_STRIDE 40   // 32 halves + 8 pad -> 80B row pitch, conflict-free frag reads

#if __has_builtin(__builtin_amdgcn_global_load_async_to_lds_b128)
#define USE_ASYNC_LDS 1
#endif

// Copy one 16B chunk global -> LDS (async DMA path on gfx1250).
__device__ __forceinline__ void stage_chunk(_Float16* l, const _Float16* g)
{
#ifdef USE_ASYNC_LDS
    __builtin_amdgcn_global_load_async_to_lds_b128(
        (__attribute__((address_space(1))) v4i*)g,
        (__attribute__((address_space(3))) v4i*)l, 0, 0);
#else
    *(float4*)l = *(const float4*)g;
#endif
}

// Stage an nrows x 32-half tile (gsrc row pitch = ld halves) into LDS.
// nrows*4 chunks of 16B spread over 128 threads.
template <int NROWS>
__device__ __forceinline__ void stage_tile(_Float16* __restrict__ ldst,
                                           const _Float16* __restrict__ gsrc, int ld)
{
#pragma unroll
    for (int i = 0; i < NROWS * 4 / 128; ++i) {
        const int c = threadIdx.x + i * 128;
        stage_chunk(ldst + (c >> 2) * LDS_STRIDE + (c & 3) * 8,
                    gsrc + (size_t)(c >> 2) * ld + (c & 3) * 8);
    }
}

// End-of-stage synchronization: drain async copies, then workgroup barrier.
__device__ __forceinline__ void stage_sync()
{
#ifdef USE_ASYNC_LDS
#if __has_builtin(__builtin_amdgcn_s_wait_asynccnt)
    __builtin_amdgcn_s_wait_asynccnt(0);
#else
    asm volatile("s_wait_asynccnt 0" ::: "memory");
#endif
#endif
    __syncthreads();
}

// Load a 16x32 f16 A/B fragment from an LDS tile row (CDNA5 ISA 7.12.2 layouts).
__device__ __forceinline__ v16h frag_a(const _Float16* base, int row, int hi)
{
    const _Float16* p = base + row * LDS_STRIDE + hi * 8;
    v8h x0 = *(const v8h*)p;
    v8h x1 = *(const v8h*)(p + 16);
    return __builtin_shufflevector(x0, x1, 0, 1, 2, 3, 4, 5, 6, 7,
                                   8, 9, 10, 11, 12, 13, 14, 15);
}
__device__ __forceinline__ v16h frag_b(const _Float16* base, int row, int hi)
{
    const _Float16* p = base + row * LDS_STRIDE + hi * 8;
    v8h x0 = *(const v8h*)p;
    v8h x1 = *(const v8h*)(p + 16);
    return __builtin_shufflevector(x0, x1, 0, 1, 2, 3, 4, 5, 6, 7,
                                   8, 9, 10, 11, 12, 13, 14, 15);
}

// ---------------------------------------------------------------------------
// Generic WMMA GEMM:  D[m][n] = sum_k A[m][k] * B[n][k]   (B given row-major N x K)
// Epilogue: +bias[n], optional exact GELU, optional +residual, write f32 and/or f16.
// Block = 128 threads = 4 waves; block tile 128x64; wave tile 32x64:
// 2 A-fragments x 4 B-fragments = 8 WMMAs per 32-wide K-step, each B fragment
// reused twice, each A fragment four times.  Operand tiles double-buffered in
// LDS via async global->LDS copies.
// All calls use M%128==0, Ncols%64==0, K%32==0 (no partial tiles -> barriers safe).
// ---------------------------------------------------------------------------
__global__ __launch_bounds__(128)
void wmma_gemm_bt(const _Float16* __restrict__ A, const _Float16* __restrict__ B,
                  const float* __restrict__ bias, const float* __restrict__ residual,
                  float* __restrict__ outf, _Float16* __restrict__ outh,
                  int M, int Ncols, int K, int lda, int ldb, int ldd,
                  int col_off, int act)
{
    __shared__ alignas(16) _Float16 lA[2][128 * LDS_STRIDE];
    __shared__ alignas(16) _Float16 lB[2][64 * LDS_STRIDE];

    const int wave = threadIdx.x >> 5;
    const int lane = threadIdx.x & 31;
    const int hi   = lane >> 4;     // half-wave select
    const int lo   = lane & 15;

    const int row0 = blockIdx.y * 128;
    const int col0 = blockIdx.x * 64;

    const _Float16* Ag = A + (size_t)row0 * lda;
    const _Float16* Bg = B + (size_t)col0 * ldb;

    v8f acc[2][4];
#pragma unroll
    for (int s = 0; s < 2; ++s)
#pragma unroll
        for (int t = 0; t < 4; ++t)
            acc[s][t] = (v8f){0.f, 0.f, 0.f, 0.f, 0.f, 0.f, 0.f, 0.f};

    // prologue: stage first K-tile
    stage_tile<128>(lA[0], Ag, lda);
    stage_tile<64>(lB[0], Bg, ldb);
    stage_sync();

    int buf = 0;
    for (int k0 = 0; k0 < K; k0 += 32) {
        // prefetch next K-tile into the other buffer while we compute
        if (k0 + 32 < K) {
            stage_tile<128>(lA[buf ^ 1], Ag + k0 + 32, lda);
            stage_tile<64>(lB[buf ^ 1], Bg + k0 + 32, ldb);
        }

        v16h a0 = frag_a(lA[buf], wave * 32 + lo, hi);
        v16h a1 = frag_a(lA[buf], wave * 32 + 16 + lo, hi);
#pragma unroll
        for (int t = 0; t < 4; ++t) {
            v16h b = frag_b(lB[buf], t * 16 + lo, hi);
            acc[0][t] = __builtin_amdgcn_wmma_f32_16x16x32_f16(
                false, a0, false, b, (short)0, acc[0][t], false, false);
            acc[1][t] = __builtin_amdgcn_wmma_f32_16x16x32_f16(
                false, a1, false, b, (short)0, acc[1][t], false, false);
        }

        stage_sync();
        buf ^= 1;
    }

    // D layout: lane lo = col; VGPR r -> row = r + 8*hi
#pragma unroll
    for (int s = 0; s < 2; ++s) {
        const int rowW = row0 + wave * 32 + s * 16;
#pragma unroll
        for (int t = 0; t < 4; ++t) {
            const int col  = col0 + t * 16 + lo;
            const int gcol = col + col_off;
            const float bv = bias ? bias[col] : 0.f;
#pragma unroll
            for (int r = 0; r < 8; ++r) {
                const int row = rowW + r + 8 * hi;
                float v = acc[s][t][r] + bv;
                if (act == 1) v = 0.5f * v * (1.0f + erff(v * 0.70710678118654752f));
                if (residual) v += residual[(size_t)row * ldd + gcol];
                if (outf) outf[(size_t)row * ldd + gcol] = v;
                if (outh) outh[(size_t)row * ldd + gcol] = (_Float16)v;
            }
        }
    }
}

// ---------------------------------------------------------------------------
// LayerNorm over C per token; emits f16 for WMMA consumption.
// ---------------------------------------------------------------------------
__global__ __launch_bounds__(256)
void ln_kernel(const float* __restrict__ x, const float* __restrict__ g,
               const float* __restrict__ b, _Float16* __restrict__ outh, int C_)
{
    const int row = blockIdx.x;
    const float* xr = x + (size_t)row * C_;
    __shared__ float sbuf[256];

    float s = 0.f, ss = 0.f;
    for (int c = threadIdx.x; c < C_; c += 256) { float v = xr[c]; s += v; ss += v * v; }

    sbuf[threadIdx.x] = s; __syncthreads();
    for (int o = 128; o > 0; o >>= 1) { if (threadIdx.x < o) sbuf[threadIdx.x] += sbuf[threadIdx.x + o]; __syncthreads(); }
    const float mean = sbuf[0] / C_;
    __syncthreads();
    sbuf[threadIdx.x] = ss; __syncthreads();
    for (int o = 128; o > 0; o >>= 1) { if (threadIdx.x < o) sbuf[threadIdx.x] += sbuf[threadIdx.x + o]; __syncthreads(); }
    const float var  = sbuf[0] / C_ - mean * mean;
    const float rstd = rsqrtf(var + 1e-6f);

    for (int c = threadIdx.x; c < C_; c += 256)
        outh[(size_t)row * C_ + c] = (_Float16)((xr[c] - mean) * rstd * g[c] + b[c]);
}

// f32 -> f16 cast (weights)
__global__ void f2h_kernel(const float* __restrict__ in, _Float16* __restrict__ out, int n)
{
    int i = blockIdx.x * blockDim.x + threadIdx.x;
    if (i < n) out[i] = (_Float16)in[i];
}

// Split qkv (N x 3C) into per-head f16 Q, K*scale, V^T.
__global__ void pack_qkv_kernel(const float* __restrict__ qkv,
                                _Float16* __restrict__ qh, _Float16* __restrict__ kh,
                                _Float16* __restrict__ vt, float scale)
{
    int idx = blockIdx.x * blockDim.x + threadIdx.x;          // over HEADS*N*HD
    if (idx >= HEADS * TOKENS * HD) return;
    const int d = idx & 63;
    const int t = idx >> 6;
    const int n = t % TOKENS;
    const int head = t / TOKENS;
    const float* base = qkv + (size_t)n * (3 * CDIM) + head * HD + d;
    qh[idx] = (_Float16)(base[0]);
    kh[idx] = (_Float16)(base[CDIM] * scale);                 // fold 1/sqrt(d) into K
    vt[(size_t)head * TOKENS * HD + (size_t)d * TOKENS + n] = (_Float16)(base[2 * CDIM]);
}

// Decomposed relative position: rel[head][ih][iw][k] = sum_c q[head][ih*48+iw][c] * tab[idx(k)][c]
// mode 0: idx = ih-k+47 (height table)  mode 1: idx = iw-k+47 (width table)
__global__ __launch_bounds__(256)
void rel_kernel(const _Float16* __restrict__ qh, const float* __restrict__ tab,
                float* __restrict__ out, int mode)
{
    const int head = blockIdx.x / GRID_HW;
    const int ih   = blockIdx.x % GRID_HW;
    for (int t = threadIdx.x; t < GRID_HW * GRID_HW; t += 256) {
        const int iw = t / GRID_HW;
        const int kk = t % GRID_HW;
        const int ridx = (mode == 0 ? ih - kk : iw - kk) + (GRID_HW - 1);
        const _Float16* qp = qh + ((size_t)head * TOKENS + ih * GRID_HW + iw) * HD;
        const float* rp = tab + (size_t)ridx * HD;
        float acc = 0.f;
#pragma unroll 8
        for (int c = 0; c < HD; ++c) acc += (float)qp[c] * rp[c];
        out[((size_t)blockIdx.x * GRID_HW + iw) * GRID_HW + kk] = acc;
    }
}

// Row softmax with decomposed rel-pos bias; emits f16 probabilities.
// Row length 2304 = 9 * 256 -> 9 values per thread, kept in registers.
__global__ __launch_bounds__(256)
void softmax_kernel(const float* __restrict__ S, const float* __restrict__ relh,
                    const float* __restrict__ relw, _Float16* __restrict__ P, int head)
{
    const int row = blockIdx.x;                               // 0..2303
    const float* bh = relh + ((size_t)head * TOKENS + row) * GRID_HW;
    const float* bw = relw + ((size_t)head * TOKENS + row) * GRID_HW;
    const float* sr = S + (size_t)row * TOKENS;
    __shared__ float sbuf[256];

    float vals[9];
    float mx = -1e30f;
#pragma unroll
    for (int i = 0; i < 9; ++i) {
        const int j = threadIdx.x + i * 256;
        const float v = sr[j] + bh[j / GRID_HW] + bw[j % GRID_HW];
        vals[i] = v;
        mx = fmaxf(mx, v);
    }
    sbuf[threadIdx.x] = mx; __syncthreads();
    for (int o = 128; o > 0; o >>= 1) { if (threadIdx.x < o) sbuf[threadIdx.x] = fmaxf(sbuf[threadIdx.x], sbuf[threadIdx.x + o]); __syncthreads(); }
    mx = sbuf[0]; __syncthreads();

    float se = 0.f;
#pragma unroll
    for (int i = 0; i < 9; ++i) { vals[i] = expf(vals[i] - mx); se += vals[i]; }
    sbuf[threadIdx.x] = se; __syncthreads();
    for (int o = 128; o > 0; o >>= 1) { if (threadIdx.x < o) sbuf[threadIdx.x] += sbuf[threadIdx.x + o]; __syncthreads(); }
    const float inv = 1.f / sbuf[0];

#pragma unroll
    for (int i = 0; i < 9; ++i)
        P[(size_t)row * TOKENS + threadIdx.x + i * 256] = (_Float16)(vals[i] * inv);
}

// ---------------------------------------------------------------------------
// Host-side orchestration (all launches on `stream`, graph-capture safe).
// ---------------------------------------------------------------------------
extern "C" void kernel_launch(void* const* d_in, const int* in_sizes, int n_in,
                              void* d_out, int out_size, void* d_ws, size_t ws_size,
                              hipStream_t stream)
{
    const float* x      = (const float*)d_in[0];   // (1,48,48,768)
    const float* w_qkv  = (const float*)d_in[1];
    const float* b_qkv  = (const float*)d_in[2];
    const float* w_proj = (const float*)d_in[3];
    const float* b_proj = (const float*)d_in[4];
    const float* rph    = (const float*)d_in[5];   // (95,64)
    const float* rpw    = (const float*)d_in[6];
    const float* ln1g   = (const float*)d_in[7];
    const float* ln1b   = (const float*)d_in[8];
    const float* ln2g   = (const float*)d_in[9];
    const float* ln2b   = (const float*)d_in[10];
    const float* w_fc1  = (const float*)d_in[11];
    const float* b_fc1  = (const float*)d_in[12];
    const float* w_fc2  = (const float*)d_in[13];
    const float* b_fc2  = (const float*)d_in[14];
    float* out = (float*)d_out;

    // ---- workspace layout (all offsets 256B aligned) ----
    char* ws = (char*)d_ws;
    size_t off = 0;
    float*    qkvS = (float*)(ws + off);    off += (size_t)TOKENS * TOKENS * 4;       // N*3C f32, reused as score matrix S (same size)
    _Float16* wbuf = (_Float16*)(ws + off); off += (size_t)MLPH * CDIM * 2;           // largest weight in f16
    _Float16* xh   = (_Float16*)(ws + off); off += (size_t)TOKENS * CDIM * 2;         // LN output (reused ln1/ln2)
    _Float16* qh   = (_Float16*)(ws + off); off += (size_t)HEADS * TOKENS * HD * 2;
    _Float16* kh   = (_Float16*)(ws + off); off += (size_t)HEADS * TOKENS * HD * 2;
    _Float16* vt   = (_Float16*)(ws + off); off += (size_t)HEADS * TOKENS * HD * 2;   // V transposed per head
    float*    relh = (float*)(ws + off);    off += (size_t)HEADS * TOKENS * GRID_HW * 4;
    float*    relw = (float*)(ws + off);    off += (size_t)HEADS * TOKENS * GRID_HW * 4;
    _Float16* Ph   = (_Float16*)(ws + off); off += (size_t)TOKENS * TOKENS * 2;       // probs (per head, reused)
    _Float16* oh   = (_Float16*)(ws + off); off += (size_t)TOKENS * CDIM * 2;         // attention output f16
    float*    attn = (float*)(ws + off);    off += (size_t)TOKENS * CDIM * 4;         // proj + residual (f32)
    _Float16* hh   = (_Float16*)(ws + off); off += (size_t)TOKENS * MLPH * 2;         // GELU(fc1) f16

    const dim3 blkG(128);
    auto cdiv = [](int a, int b) { return (a + b - 1) / b; };

    // 1) LN1 -> xh (f16)
    ln_kernel<<<TOKENS, 256, 0, stream>>>(x, ln1g, ln1b, xh, CDIM);

    // 2) QKV projection: (2304x768) x (2304x768)^T -> qkvS (f32, ld=2304)
    f2h_kernel<<<cdiv(3 * CDIM * CDIM, 256), 256, 0, stream>>>(w_qkv, wbuf, 3 * CDIM * CDIM);
    wmma_gemm_bt<<<dim3(3 * CDIM / 64, TOKENS / 128), blkG, 0, stream>>>(
        xh, wbuf, b_qkv, nullptr, qkvS, nullptr,
        TOKENS, 3 * CDIM, CDIM, CDIM, CDIM, 3 * CDIM, 0, 0);

    // 3) split heads: Q, K*scale, V^T (f16)
    pack_qkv_kernel<<<cdiv(HEADS * TOKENS * HD, 256), 256, 0, stream>>>(
        qkvS, qh, kh, vt, 0.125f /* 64^-0.5 */);

    // 4) decomposed rel-pos tables (use raw Q)
    rel_kernel<<<HEADS * GRID_HW, 256, 0, stream>>>(qh, rph, relh, 0);
    rel_kernel<<<HEADS * GRID_HW, 256, 0, stream>>>(qh, rpw, relw, 1);

    // 5) per-head attention; S reuses the (now dead) qkvS region -> stays L2 resident
    float* S = qkvS;
    for (int head = 0; head < HEADS; ++head) {
        const _Float16* qhh = qh + (size_t)head * TOKENS * HD;
        const _Float16* khh = kh + (size_t)head * TOKENS * HD;
        const _Float16* vth = vt + (size_t)head * TOKENS * HD;
        // scores: Q (2304x64) x K^T
        wmma_gemm_bt<<<dim3(TOKENS / 64, TOKENS / 128), blkG, 0, stream>>>(
            qhh, khh, nullptr, nullptr, S, nullptr,
            TOKENS, TOKENS, HD, HD, HD, TOKENS, 0, 0);
        // +bias, softmax -> Ph (f16)
        softmax_kernel<<<TOKENS, 256, 0, stream>>>(S, relh, relw, Ph, head);
        // P (2304x2304) x V  (B = V^T row-major 64x2304) -> oh[:, head*64:...]
        wmma_gemm_bt<<<dim3(HD / 64, TOKENS / 128), blkG, 0, stream>>>(
            Ph, vth, nullptr, nullptr, nullptr, oh,
            TOKENS, HD, TOKENS, TOKENS, TOKENS, CDIM, head * HD, 0);
    }

    // 6) output projection + residual(x) -> attn (f32)
    f2h_kernel<<<cdiv(CDIM * CDIM, 256), 256, 0, stream>>>(w_proj, wbuf, CDIM * CDIM);
    wmma_gemm_bt<<<dim3(CDIM / 64, TOKENS / 128), blkG, 0, stream>>>(
        oh, wbuf, b_proj, x, attn, nullptr,
        TOKENS, CDIM, CDIM, CDIM, CDIM, CDIM, 0, 0);

    // 7) LN2 -> xh (f16)
    ln_kernel<<<TOKENS, 256, 0, stream>>>(attn, ln2g, ln2b, xh, CDIM);

    // 8) fc1 + exact GELU -> hh (f16)
    f2h_kernel<<<cdiv(MLPH * CDIM, 256), 256, 0, stream>>>(w_fc1, wbuf, MLPH * CDIM);
    wmma_gemm_bt<<<dim3(MLPH / 64, TOKENS / 128), blkG, 0, stream>>>(
        xh, wbuf, b_fc1, nullptr, nullptr, hh,
        TOKENS, MLPH, CDIM, CDIM, CDIM, MLPH, 0, 1 /* gelu */);

    // 9) fc2 + residual(attn) -> out (f32)
    f2h_kernel<<<cdiv(CDIM * MLPH, 256), 256, 0, stream>>>(w_fc2, wbuf, CDIM * MLPH);
    wmma_gemm_bt<<<dim3(CDIM / 64, TOKENS / 128), blkG, 0, stream>>>(
        hh, wbuf, b_fc2, attn, out, nullptr,
        TOKENS, CDIM, MLPH, MLPH, MLPH, CDIM, 0, 0);
}
